// LSTMLayer_12249246728587
// MI455X (gfx1250) — matrix operand order repaired and checked by reading, not compiled, not run on hardware
//
#include <hip/hip_runtime.h>
#include <hip/hip_bf16.h>

// Problem constants (match reference): T=256, B=128, I=H=1024
#define TS   256
#define BS   128
#define INP  1024
#define HID  1024
#define G4   (4 * HID)

typedef __bf16 v16bf __attribute__((ext_vector_type(16)));
typedef __bf16 v8bf  __attribute__((ext_vector_type(8)));
typedef float  v8f   __attribute__((ext_vector_type(8)));

__device__ __forceinline__ unsigned short f2bf(float f) {
    unsigned int u = __float_as_uint(f);
    u += 0x7FFFu + ((u >> 16) & 1u);          // round-to-nearest-even
    return (unsigned short)(u >> 16);
}

__device__ __forceinline__ v8f zero8() {
    v8f z = {0.f, 0.f, 0.f, 0.f, 0.f, 0.f, 0.f, 0.f};
    return z;
}

// ---- WMMA fragment loaders (wave32, 16x16x32 bf16) -------------------------
// A (MxK = 16x32), row-major source with row stride `ld` (elements).
// Lane L holds row M = L%16; lanes 0-15: K = k0+[0..15];
// lanes 16-31: K = k0+[16..31].
__device__ __forceinline__ v16bf load_a_frag(const unsigned short* __restrict__ base,
                                             int ld, int k0, int lane) {
    int m    = lane & 15;
    int ksel = (lane >> 4) << 4;              // 0 or 16
    const unsigned short* p = base + (size_t)m * ld + k0 + ksel;
    return *reinterpret_cast<const v16bf*>(p);
}

// B (KxN = 32x16). We compute  out = act @ W.T, so column n of B == row n of
// row-major W. Lane L holds column n = L%16; lanes 0-15: K = k0+[0..15],
// lanes 16-31: K = k0+[16..31]  -> one contiguous 32-byte load per lane.
__device__ __forceinline__ v16bf load_b_frag(const unsigned short* __restrict__ w,
                                             int ld, int nrow0, int k0, int lane) {
    int n  = lane & 15;
    int kk = k0 + ((lane >> 4) << 4);         // +16 for upper half-wave
    return *reinterpret_cast<const v16bf*>(w + (size_t)(nrow0 + n) * ld + kk);
}

// ---- helpers ----------------------------------------------------------------
__global__ void cvt_f32_to_bf16(const float* __restrict__ src,
                                unsigned short* __restrict__ dst, int n) {
    int i = blockIdx.x * blockDim.x + threadIdx.x;
    int stride = gridDim.x * blockDim.x;
    for (; i < n; i += stride) dst[i] = f2bf(src[i]);
}

__global__ void copy_f32(const float* __restrict__ src, float* __restrict__ dst, int n) {
    int i = blockIdx.x * blockDim.x + threadIdx.x;
    int stride = gridDim.x * blockDim.x;
    for (; i < n; i += stride) dst[i] = src[i];
}

// ---- Phase 1: pre = x @ Wih^T + (b_ih + b_hh)  (fully parallel big GEMM) ----
// Register-blocked: each wave computes a 32x64 output block (2 M-tiles x
// 4 N-tiles) -> 8 WMMAs per 6 fragment loads.
// x_bf: [TS*BS, INP] bf16; wih: [4H, INP] bf16 row-major; pre: [TS*BS, 4H] f32.
__global__ __launch_bounds__(256) void pregemm_kernel(
        const unsigned short* __restrict__ x_bf,
        const unsigned short* __restrict__ wih,
        const float* __restrict__ b_ih,
        const float* __restrict__ b_hh,
        float* __restrict__ pre) {
    int lane = threadIdx.x & 31;
    int wave = threadIdx.x >> 5;
    int gid  = blockIdx.x * 8 + wave;         // 0 .. 1024*64-1
    int ng   = gid & 63;                      // 64 N-groups of 4 tiles (4H/64)
    int mg   = gid >> 6;                      // 1024 M-groups of 2 tiles (TS*BS/32)

    const unsigned short* abase = x_bf + (size_t)mg * 32 * INP;
    int ncol0 = ng * 64;

    v8f acc[2][4];
#pragma unroll
    for (int mi = 0; mi < 2; ++mi)
#pragma unroll
        for (int j = 0; j < 4; ++j) acc[mi][j] = zero8();

    for (int k0 = 0; k0 < INP; k0 += 32) {
        v16bf a0 = load_a_frag(abase,            INP, k0, lane);
        v16bf a1 = load_a_frag(abase + 16 * INP, INP, k0, lane);
#pragma unroll
        for (int j = 0; j < 4; ++j) {
            v16bf b = load_b_frag(wih, INP, ncol0 + j * 16, k0, lane);
            acc[0][j] = __builtin_amdgcn_wmma_f32_16x16x32_bf16(
                            false, a0, false, b, (short)0, acc[0][j], false, false);
            acc[1][j] = __builtin_amdgcn_wmma_f32_16x16x32_bf16(
                            false, a1, false, b, (short)0, acc[1][j], false, false);
        }
    }

    int n    = lane & 15;
    int half = lane >> 4;
#pragma unroll
    for (int j = 0; j < 4; ++j) {
        int col  = ncol0 + j * 16 + n;
        float bias = b_ih[col] + b_hh[col];
#pragma unroll
        for (int mi = 0; mi < 2; ++mi) {
#pragma unroll
            for (int r = 0; r < 8; ++r) {
                int row = mg * 32 + mi * 16 + r + half * 8;  // C layout: VGPR r -> M=r / r+8
                pre[(size_t)row * G4 + col] = acc[mi][j][r] + bias;
            }
        }
    }
}

// ---- Phase 2: one timestep --------------------------------------------------
// Each wave: 2 batch-tiles x 16 hidden cols, all 4 gates fused (8 accumulators,
// A frags reused 4x, B frags reused 2x). Epilogue does the cell update per lane.
__global__ __launch_bounds__(128) void lstm_step_kernel(
        const unsigned short* __restrict__ whh,   // [4H, HID] bf16
        const float* __restrict__ pre_t,          // [BS, 4H] this step's x-part
        const unsigned short* __restrict__ h_prev,// [BS, HID] bf16
        unsigned short* __restrict__ h_next,      // [BS, HID] bf16
        float* __restrict__ cbuf,                 // [BS, HID] f32 (in/out)
        float* __restrict__ ys_t) {               // [BS, HID] f32 out
    int lane = threadIdx.x & 31;
    int wave = threadIdx.x >> 5;
    int mg   = blockIdx.x & 3;                    // 4 batch groups (2 tiles each)
    int hb   = (blockIdx.x >> 2) * 4 + wave;      // 64 hidden tiles

    const unsigned short* abase = h_prev + (size_t)mg * 32 * HID;
    int hcol0 = hb * 16;

    v8f acc[2][4];                                // [batch tile][gate]
#pragma unroll
    for (int mi = 0; mi < 2; ++mi)
#pragma unroll
        for (int g = 0; g < 4; ++g) acc[mi][g] = zero8();

    for (int k0 = 0; k0 < HID; k0 += 32) {
        v16bf a0 = load_a_frag(abase,            HID, k0, lane);
        v16bf a1 = load_a_frag(abase + 16 * HID, HID, k0, lane);
#pragma unroll
        for (int g = 0; g < 4; ++g) {
            v16bf b = load_b_frag(whh, HID, g * HID + hcol0, k0, lane);
            acc[0][g] = __builtin_amdgcn_wmma_f32_16x16x32_bf16(
                            false, a0, false, b, (short)0, acc[0][g], false, false);
            acc[1][g] = __builtin_amdgcn_wmma_f32_16x16x32_bf16(
                            false, a1, false, b, (short)0, acc[1][g], false, false);
        }
    }

    int n    = lane & 15;
    int half = lane >> 4;
    int col  = hcol0 + n;
#pragma unroll
    for (int mi = 0; mi < 2; ++mi) {
#pragma unroll
        for (int r = 0; r < 8; ++r) {
            int brow = mg * 32 + mi * 16 + r + half * 8;   // batch row
            size_t pb = (size_t)brow * G4 + col;
            float gi = acc[mi][0][r] + pre_t[pb + 0 * HID];
            float gf = acc[mi][1][r] + pre_t[pb + 1 * HID];
            float gg = acc[mi][2][r] + pre_t[pb + 2 * HID];
            float go = acc[mi][3][r] + pre_t[pb + 3 * HID];
            gi = 1.f / (1.f + __expf(-gi));
            gf = 1.f / (1.f + __expf(-gf));
            gg = tanhf(gg);
            go = 1.f / (1.f + __expf(-go));
            size_t hp = (size_t)brow * HID + col;
            float cn = gf * cbuf[hp] + gi * gg;
            float hn = go * tanhf(cn);
            cbuf[hp]   = cn;
            ys_t[hp]   = hn;
            h_next[hp] = f2bf(hn);
        }
    }
}

// ---- Phase 3: emit hT, cT ---------------------------------------------------
__global__ void finalize_kernel(const float* __restrict__ last_h,
                                const float* __restrict__ cbuf,
                                float* __restrict__ out_h,
                                float* __restrict__ out_c, int n) {
    int i = blockIdx.x * blockDim.x + threadIdx.x;
    int stride = gridDim.x * blockDim.x;
    for (; i < n; i += stride) {
        out_h[i] = last_h[i];
        out_c[i] = cbuf[i];
    }
}

extern "C" void kernel_launch(void* const* d_in, const int* in_sizes, int n_in,
                              void* d_out, int out_size, void* d_ws, size_t ws_size,
                              hipStream_t stream) {
    const float* x    = (const float*)d_in[0];   // [TS, BS, INP]
    const float* h0   = (const float*)d_in[1];   // [BS, HID]
    const float* c0   = (const float*)d_in[2];   // [BS, HID]
    const float* W_ih = (const float*)d_in[3];   // [4H, INP]
    const float* W_hh = (const float*)d_in[4];   // [4H, HID]
    const float* b_ih = (const float*)d_in[5];   // [4H]
    const float* b_hh = (const float*)d_in[6];   // [4H]
    float* out = (float*)d_out;                  // ys | hT | cT

    // workspace carve-up (256B aligned)
    char*  ws  = (char*)d_ws;
    size_t off = 0;
    auto carve = [&](size_t bytes) -> void* {
        void* p = ws + off;
        off = (off + bytes + 255) & ~(size_t)255;
        return p;
    };
    unsigned short* wih_bf = (unsigned short*)carve((size_t)G4 * INP * 2);      // 8 MB
    unsigned short* whh_bf = (unsigned short*)carve((size_t)G4 * HID * 2);      // 8 MB
    unsigned short* x_bf   = (unsigned short*)carve((size_t)TS * BS * INP * 2); // 64 MB
    unsigned short* hbufA  = (unsigned short*)carve((size_t)BS * HID * 2);
    unsigned short* hbufB  = (unsigned short*)carve((size_t)BS * HID * 2);
    float*          cbuf   = (float*)carve((size_t)BS * HID * 4);
    float*          pre    = (float*)carve((size_t)TS * BS * G4 * 4);           // 512 MB

    // 0) convert weights / x / h0 to bf16, copy c0
    cvt_f32_to_bf16<<<1024, 256, 0, stream>>>(W_ih, wih_bf, G4 * INP);
    cvt_f32_to_bf16<<<1024, 256, 0, stream>>>(W_hh, whh_bf, G4 * HID);
    cvt_f32_to_bf16<<<4096, 256, 0, stream>>>(x, x_bf, TS * BS * INP);
    cvt_f32_to_bf16<<<512,  256, 0, stream>>>(h0, hbufA, BS * HID);
    copy_f32<<<512, 256, 0, stream>>>(c0, cbuf, BS * HID);

    // 1) pre = x @ Wih^T + bias   (1024 M-groups x 64 N-groups, 8 waves/block)
    pregemm_kernel<<<(1024 * 64) / 8, 256, 0, stream>>>(x_bf, wih_bf, b_ih, b_hh, pre);

    // 2) recurrence: one launch per timestep, double-buffered h
    for (int t = 0; t < TS; ++t) {
        const unsigned short* hp = (t & 1) ? hbufB : hbufA;
        unsigned short*       hn = (t & 1) ? hbufA : hbufB;
        lstm_step_kernel<<<64, 128, 0, stream>>>(
            whh_bf, pre + (size_t)t * BS * G4, hp, hn, cbuf,
            out + (size_t)t * BS * HID);
    }

    // 3) hT, cT tails
    finalize_kernel<<<512, 256, 0, stream>>>(
        out + (size_t)(TS - 1) * BS * HID, cbuf,
        out + (size_t)TS * BS * HID,
        out + (size_t)TS * BS * HID + (size_t)BS * HID,
        BS * HID);
}
